// ISTFT_26388279067174
// MI455X (gfx1250) — matrix operand order repaired and checked by reading, not compiled, not run on hardware
//
#include <hip/hip_runtime.h>

typedef __attribute__((ext_vector_type(16))) _Float16 v16h;
typedef __attribute__((ext_vector_type(8)))  _Float16 v8h;
typedef __attribute__((ext_vector_type(4)))  _Float16 v4h;
typedef __attribute__((ext_vector_type(8)))  float    v8f;
typedef __attribute__((ext_vector_type(4)))  float    v4f;

#define NFREQ   513
#define TFRM    2048
#define NFFT    1024
#define HOP     256
#define NBATCH  16
#define KKH     544      // padded per-matrix K (513 -> 544, multiple of 32)
#define KKT     1088     // A row stride in halves
#define KKE     1056     // effective GEMM K (last all-zero chunk dropped)
#define NCHUNK  33       // KKE / 32
#define OUTFULL 525056   // (TFRM-1)*HOP + NFFT
#define OUTP    524288   // OUTFULL - 2*PAD  (= 2^19)
#define PAD     384
#define TM      128      // n-tile per pass
#define TN      128      // t-tile per block
#define BSTR    1064     // LDS panel row stride in halves (2128B: 16B-aligned, conflict-free frags)
#define A_BYTES ((size_t)NFFT * KKT * 2)   // 2,228,224 B

// ---------------- twiddle/window prep: A[n][kk] in f16 -----------------------
__global__ __launch_bounds__(256) void prep_twiddle(const float* __restrict__ win,
                                                    _Float16* __restrict__ Ag) {
  int idx = blockIdx.x * 256 + threadIdx.x;
  if (idx >= NFFT * KKT) return;
  int n = idx / KKT, kk = idx % KKT;
  const float step = 6.283185307179586f / 1024.f;
  float coef = 0.f;
  if (kk < KKH) {                       // cosine half (spec_real)
    int k = kk;
    if (k < NFREQ) {
      float cw = (k == 0 || k == NFFT / 2) ? 1.f : 2.f;
      int ph = (n * k) & (NFFT - 1);    // exact integer phase reduction
      coef = cw * __cosf((float)ph * step);
    }
  } else {                              // sine half (spec_imag)
    int k = kk - KKH;
    if (k >= 1 && k < NFFT / 2) {
      int ph = (n * k) & (NFFT - 1);
      coef = -2.f * __sinf((float)ph * step);
    }
  }
  Ag[idx] = (_Float16)(coef * win[n]);  // window folded in; 1/1024 applied later
}

// ---------------- zero the overlap-add accumulator ---------------------------
__global__ __launch_bounds__(256) void zerok(float* __restrict__ p, int n) {
  int i = blockIdx.x * 256 + threadIdx.x;
  if (i < n) p[i] = 0.f;
}

// ------- native f32 global atomic add with immediate offset (no CAS loop) ----
__device__ __forceinline__ void gadd_f32_off(float* base, int imm, float v) {
  asm volatile("global_atomic_add_f32 %0, %1, off offset:%2"
               : : "v"(base), "v"(v), "n"(imm) : "memory");
}

// ---------------- fused irfft-GEMM + window + overlap-add --------------------
// One block owns a 128-t column for one batch: stages the full K-panel of both
// spectra (f16) into LDS once (266 KB of the 320 KB WGP LDS), then sweeps all
// 8 n-tiles without further barriers. Spec is read from HBM exactly once.
__global__ __launch_bounds__(256) void istft_gemm_oa(
    const float* __restrict__ specR, const float* __restrict__ specI,
    const _Float16* __restrict__ Ag, float* __restrict__ acc) {
  __shared__ __align__(16) _Float16 Bs[TN * BSTR];   // 272,384 B

  const int tid   = threadIdx.x;
  const int lane  = tid & 31;
  const int wid   = tid >> 5;      // 8 waves
  const int waveM = wid >> 2;      // 0..1  -> 64 rows (n) each
  const int waveN = wid & 3;       // 0..3  -> 32 cols (t) each
  const int g     = lane >> 4;     // half-wave group
  const int l16   = lane & 15;

  const int tBase = blockIdx.x * TN;
  const int b     = blockIdx.y;

  // ---- Stage full B panel: f32 spec -> f16 LDS [t][kk], 4x4 tiles per thread.
  // Address clamp (not value mask): A holds zeros in all padded columns, so
  // clamped duplicate rows contribute nothing. No exec divergence.
#pragma unroll 1
  for (int it = 0; it < 33; ++it) {
    int w  = it * 256 + tid;        // 0..8447
    int tq = w & 31;                // t quad (coalesced across lanes)
    int kq = w >> 5;                // 0..263 (uniform per wave)
    int kk0q = kq * 4;
    const float* __restrict__ src = (kk0q < KKH) ? specR : specI;
    int kb = (kk0q < KKH) ? kk0q : (kk0q - KKH);
    v4f f[4];
#pragma unroll
    for (int j = 0; j < 4; ++j) {
      int kg = kb + j;
      kg = (kg > NFREQ - 1) ? (NFREQ - 1) : kg;   // address clamp only
      f[j] = *(const v4f*)&src[((size_t)(b * NFREQ + kg)) * TFRM + tBase + tq * 4];
    }
#pragma unroll
    for (int tt = 0; tt < 4; ++tt) {
      v4h pk;
      pk[0] = (_Float16)f[0][tt];
      pk[1] = (_Float16)f[1][tt];
      pk[2] = (_Float16)f[2][tt];
      pk[3] = (_Float16)f[3][tt];
      *(v4h*)&Bs[(size_t)(tq * 4 + tt) * BSTR + kk0q] = pk;   // ds_store_b64
    }
  }
  __syncthreads();   // only barrier in the kernel; panel is read-only below

  // ---- Sweep the 8 n-tiles against the resident panel ----
#pragma unroll 1
  for (int nt = 0; nt < NFFT / TM; ++nt) {
    const int nBase = nt * TM;
    v8f c[4][2] = {};

#pragma unroll 1
    for (int chunk = 0; chunk < NCHUNK; ++chunk) {
      const int kk0 = chunk * 32;

      // A fragments from global (2.2 MB table, L2-resident)
      v16h af[4];
#pragma unroll
      for (int mf = 0; mf < 4; ++mf) {
        const _Float16* ap =
            Ag + (size_t)(nBase + waveM * 64 + mf * 16 + l16) * KKT + kk0 + g * 8;
        v8h lo = *(const v8h*)ap;          // K = g*8 .. g*8+7
        v8h hi = *(const v8h*)(ap + 16);   // K = 16+g*8 .. 16+g*8+7
        af[mf] = __builtin_shufflevector(lo, hi, 0, 1, 2, 3, 4, 5, 6, 7,
                                         8, 9, 10, 11, 12, 13, 14, 15);
      }
      // B fragments from LDS panel: lane column t = l16, K = g*8 .. g*8+7, 16+g*8 .. 16+g*8+7
      v16h bf[2];
#pragma unroll
      for (int nf = 0; nf < 2; ++nf) {
        const _Float16* bp =
            &Bs[(size_t)(waveN * 32 + nf * 16 + l16) * BSTR + kk0 + g * 8];
        v8h lo = *(const v8h*)bp;
        v8h hi = *(const v8h*)(bp + 16);
        bf[nf] = __builtin_shufflevector(lo, hi, 0, 1, 2, 3, 4, 5, 6, 7,
                                         8, 9, 10, 11, 12, 13, 14, 15);
      }
#pragma unroll
      for (int mf = 0; mf < 4; ++mf)
#pragma unroll
        for (int nf = 0; nf < 2; ++nf)
          c[mf][nf] = __builtin_amdgcn_wmma_f32_16x16x32_f16(
              false, af[mf], false, bf[nf], (short)0, c[mf][nf], false, false);
    }

    // Overlap-add epilogue: frame[n,t] -> y[b, t*HOP + n].
    // One base address per lane; all 64 element offsets are compile-time
    // immediates inside global_atomic_add_f32's 24-bit IOFFSET field.
    float* basep = &acc[(size_t)b * OUTFULL
                        + (size_t)(tBase + waveN * 32 + l16) * HOP
                        + (nBase + waveM * 64 + g * 8)];
#pragma unroll
    for (int mf = 0; mf < 4; ++mf)
#pragma unroll
      for (int nf = 0; nf < 2; ++nf)
#pragma unroll
        for (int v = 0; v < 8; ++v) {
          float val = c[mf][nf][v];
          gadd_f32_off(basep, (nf * 16 * HOP + mf * 16 + v) * 4, val);
        }
  }
}

// ---------------- envelope normalize + trim ----------------------------------
__global__ __launch_bounds__(256) void finalize(const float* __restrict__ acc,
                                                const float* __restrict__ win,
                                                float* __restrict__ out) {
  int i = blockIdx.x * 256 + threadIdx.x;
  if (i >= NBATCH * OUTP) return;
  int b = i >> 19;                 // OUTP == 2^19
  int p = i & (OUTP - 1);
  int s = p + PAD;
  int q = s >> 8, rem = s & 255;
  float env = 0.f;
#pragma unroll
  for (int r = 0; r < 4; ++r) {    // the 4 frames covering sample s
    int t = q - r;
    if (t >= 0 && t < TFRM) {
      float w = win[rem + r * 256];
      env += w * w;
    }
  }
  out[i] = acc[(size_t)b * OUTFULL + s] * (1.f / 1024.f) / (env + 1e-11f);
}

extern "C" void kernel_launch(void* const* d_in, const int* in_sizes, int n_in,
                              void* d_out, int out_size, void* d_ws, size_t ws_size,
                              hipStream_t stream) {
  const float* specR = (const float*)d_in[0];
  const float* specI = (const float*)d_in[1];
  const float* win   = (const float*)d_in[2];
  float* out = (float*)d_out;

  _Float16* Ag = (_Float16*)d_ws;
  float* acc = (float*)((char*)d_ws + A_BYTES);

  {
    int n = NFFT * KKT;
    prep_twiddle<<<(n + 255) / 256, 256, 0, stream>>>(win, Ag);
  }
  {
    int n = NBATCH * OUTFULL;
    zerok<<<(n + 255) / 256, 256, 0, stream>>>(acc, n);
  }
  {
    dim3 grid(TFRM / TN, NBATCH);   // (16, 16): one block per (t-tile, batch)
    istft_gemm_oa<<<grid, 256, 0, stream>>>(specR, specI, Ag, acc);
  }
  {
    int n = NBATCH * OUTP;
    finalize<<<n / 256, 256, 0, stream>>>(acc, win, out);
  }
}
